// ImplicitLayer_14912126452358
// MI455X (gfx1250) — matrix operand order repaired and checked by reading, not compiled, not run on hardware
//
#include <hip/hip_runtime.h>
#include <hip/hip_bf16.h>

// ---------------------------------------------------------------------------
// Implicit layer (deep equilibrium) on MI455X / gfx1250.
//
//   inj = u @ Bw^T + b_B + b_A            (GEMM, done once)
//   z   <- tanh(z @ Aw^T + inj)           (fixed-point, HBM-bound: 402MB/iter
//                                          @23.3TB/s = 17us vs ~4us of f16
//                                          WMMA compute -> memory bound)
//
// f16 WMMA (v_wmma_f32_16x16x32_f16, f32 accumulate): free vs fp8 under the
// memory roofline, accurate enough (10-bit mantissa) for the 1e-3 residual.
// Tiling: wave = 16 rows x 128 cols (8 accum tiles = 64 VGPRs -> no spills);
// block = 8 waves = 64 rows x 256 cols. Weights (128KB f16) staged into LDS
// once per block with padded row stride (conflict-free ds_load_b128).
// z updated in place (row-independent update; __syncthreads separates the
// block's z reads from its z writes since two waves share each row group).
// Convergence: deterministic two-level reduction + sticky device flag gating
// 32 pre-enqueued iteration kernels (graph-capture safe, no host sync).
// ---------------------------------------------------------------------------

typedef __attribute__((ext_vector_type(16))) _Float16 v16h;
typedef __attribute__((ext_vector_type(8)))  _Float16 v8h;
typedef __attribute__((ext_vector_type(8)))  float    v8f;
typedef __attribute__((ext_vector_type(4)))  float    v4f;

#define BATCH          131072
#define DIM            256
#define KCHUNKS        8                        // 256 / 32
#define NT_WAVE        8                        // N-tiles per wave (128 cols)
#define WAVES          8
#define ROWS_PER_WAVE  16
#define ROWS_PER_BLOCK 64                       // 4 row-groups x 2 col-halves
#define NBLOCKS        (BATCH / ROWS_PER_BLOCK) // 2048
#define NITER          32                       // >> ~5 needed (contraction ~0.1)
#define EPS_RES        1e-3f

#define LDS_ROW_H      264                      // padded halves per weight row
#define W_LDS_BYTES    (DIM * LDS_ROW_H * 2)    // 135168 B
#define RED_LDS_BYTES  2048                     // 512 floats for block reduce

// workspace layout (byte offsets)
#define WS_FLAGS   0                               // int[64] sticky converged flags
#define WS_BIAS    256                             // float[256] (A_bias + B_bias)
#define WS_PARTIAL 2048                            // float[NITER*4096] residual partials
#define WS_AWH     (2048 + NITER * 4096 * 4)       // f16[256*256] A_weight
#define WS_BWH     (WS_AWH + DIM * DIM * 2)        // f16[256*256] B_weight
#define WS_INJ     (WS_BWH + DIM * DIM * 2)        // float[BATCH*DIM] injection

__device__ __forceinline__ float fast_tanh(float x) {
  float ax = fabsf(x);
  float e  = __expf(-2.0f * ax);          // v_exp_f32 path
  float r  = (1.0f - e) / (1.0f + e);
  return copysignf(r, x);
}

// --- one-time prep: f32 -> f16 weights, fused bias, zero control state -----
__global__ void deq_prep_kernel(const float* __restrict__ Aw,
                                const float* __restrict__ Ab,
                                const float* __restrict__ Bw,
                                const float* __restrict__ Bb,
                                unsigned char* __restrict__ ws) {
  int t = blockIdx.x * blockDim.x + threadIdx.x;
  _Float16* Awh = (_Float16*)(ws + WS_AWH);
  _Float16* Bwh = (_Float16*)(ws + WS_BWH);
  if (t < DIM * DIM) {
    Awh[t] = (_Float16)Aw[t];
    Bwh[t] = (_Float16)Bw[t];
  }
  if (t < DIM) ((float*)(ws + WS_BIAS))[t] = Ab[t] + Bb[t];
  if (t < 64)  ((int*)(ws + WS_FLAGS))[t] = 0;
}

// --- stage 256x256 f16 weight matrix into LDS with padded row stride -------
__device__ __forceinline__ void stage_weights(const _Float16* __restrict__ Wg,
                                              char* __restrict__ lds) {
  const uint4* src = (const uint4*)Wg;              // 16B chunks, 32 per row
  for (int c = threadIdx.x; c < DIM * 32; c += blockDim.x) {
    int row = c >> 5, pos = c & 31;
    uint4 v = src[row * 32 + pos];
    *(uint4*)(lds + row * (LDS_ROW_H * 2) + pos * 16) = v;
  }
  __syncthreads();
}

// --- core: one 16-row x 128-col output tile per wave, K=256 ----------------
// A fragment (16x32 f16): lane = M row; lanes 0-15: halves = K {0..7,16..23},
// lanes 16-31: halves = K {8..15,24..31} (ISA 7.12.2).
// B fragment (32x16 f16): lane%16 = N col; lane<16 -> K {0..7,16..23}, else K {8..15,24..31}.
__device__ __forceinline__ void gemm_rowtile(const float* __restrict__ X,
                                             const char* __restrict__ wlds,
                                             int rowbase, int ntile0,
                                             v8f acc[NT_WAVE]) {
  const int lane = threadIdx.x & 31;
  const int half = lane >> 4, lr = lane & 15;
  const float* xrow = X + (size_t)(rowbase + lr) * DIM;
#pragma unroll
  for (int c = 0; c < KCHUNKS; ++c) {
    const int k0 = c * 32 + half * 8;
    v4f f0 = *(const v4f*)(xrow + k0);
    v4f f1 = *(const v4f*)(xrow + k0 + 4);
    v4f f2 = *(const v4f*)(xrow + k0 + 16);
    v4f f3 = *(const v4f*)(xrow + k0 + 20);
    v16h a;
#pragma unroll
    for (int q = 0; q < 4; ++q) {
      a[q]      = (_Float16)f0[q];
      a[4 + q]  = (_Float16)f1[q];
      a[8 + q]  = (_Float16)f2[q];
      a[12 + q] = (_Float16)f3[q];
    }
    const char* bbase = wlds + (c * 32 + half * 8) * 2;
#pragma unroll
    for (int n = 0; n < NT_WAVE; ++n) {
      const char* bp =
          bbase + (size_t)((ntile0 + n) * 16 + lr) * (LDS_ROW_H * 2);
      v8h blo = *(const v8h*)bp;          // ds_load_b128
      v8h bhi = *(const v8h*)(bp + 32);   // ds_load_b128
      v16h b;
#pragma unroll
      for (int q = 0; q < 8; ++q) { b[q] = blo[q]; b[8 + q] = bhi[q]; }
      acc[n] = __builtin_amdgcn_wmma_f32_16x16x32_f16(
          false, a, false, b, (short)0, acc[n], false, false);
    }
  }
}

// --- inject: inj = u @ Bw^T + bias ; z = tanh(inj)  (this IS f(z0=0)) ------
__global__ void __launch_bounds__(256)
deq_inject_kernel(const float* __restrict__ u,
                  unsigned char* __restrict__ ws,
                  float* __restrict__ z) {
  extern __shared__ char smem[];
  stage_weights((const _Float16*)(ws + WS_BWH), smem);
  const float* bias = (const float*)(ws + WS_BIAS);
  float* inj = (float*)(ws + WS_INJ);

  const int wave = threadIdx.x >> 5, lane = threadIdx.x & 31;
  const int half = lane >> 4, lr = lane & 15;
  const int rowbase = blockIdx.x * ROWS_PER_BLOCK + (wave >> 1) * ROWS_PER_WAVE;
  const int ntile0  = (wave & 1) * NT_WAVE;

  v8f acc[NT_WAVE] = {};
  gemm_rowtile(u, smem, rowbase, ntile0, acc);

#pragma unroll
  for (int n = 0; n < NT_WAVE; ++n) {
    const int col = (ntile0 + n) * 16 + lr;
    const float bj = bias[col];
#pragma unroll
    for (int r = 0; r < 8; ++r) {
      const int row = rowbase + r + half * 8;      // C/D layout: M=r / r+8
      const size_t idx = (size_t)row * DIM + col;
      const float v = acc[n][r] + bj;
      inj[idx] = v;
      z[idx] = fast_tanh(v);
    }
  }
}

// --- one fixed-point body: z <- tanh(z @ Aw^T + inj), residual partials ----
__global__ void __launch_bounds__(256)
deq_iter_kernel(unsigned char* __restrict__ ws,
                float* __restrict__ z, int k) {
  // sticky convergence gate (kernel-boundary coherent; uniform per block)
  if (((volatile const int*)(ws + WS_FLAGS))[k - 1]) return;

  extern __shared__ char smem[];
  float* red = (float*)(smem + W_LDS_BYTES);
  stage_weights((const _Float16*)(ws + WS_AWH), smem);
  const float* inj = (const float*)(ws + WS_INJ);

  const int wave = threadIdx.x >> 5, lane = threadIdx.x & 31;
  const int half = lane >> 4, lr = lane & 15;
  const int rowbase = blockIdx.x * ROWS_PER_BLOCK + (wave >> 1) * ROWS_PER_WAVE;
  const int ntile0  = (wave & 1) * NT_WAVE;

  v8f acc[NT_WAVE] = {};
  gemm_rowtile(z, smem, rowbase, ntile0, acc);

  // two waves share each 16-row group (different col halves): make sure all
  // z reads (A-fragments) in this block complete before any z write.
  __syncthreads();

  float d2 = 0.0f, n2 = 0.0f;
#pragma unroll
  for (int n = 0; n < NT_WAVE; ++n) {
    const int col = (ntile0 + n) * 16 + lr;
#pragma unroll
    for (int r = 0; r < 8; ++r) {
      const int row = rowbase + r + half * 8;
      const size_t idx = (size_t)row * DIM + col;
      const float zn = fast_tanh(acc[n][r] + inj[idx]);
      const float zo = z[idx];
      const float d = zn - zo;
      d2 += d * d;
      n2 += zn * zn;
      z[idx] = zn;                       // in-place: rows disjoint per block
    }
  }

  // deterministic block tree reduction (fixed pairing order)
  red[threadIdx.x] = d2;
  red[256 + threadIdx.x] = n2;
  __syncthreads();
  for (int s = 128; s > 0; s >>= 1) {
    if ((int)threadIdx.x < s) {
      red[threadIdx.x]       += red[threadIdx.x + s];
      red[256 + threadIdx.x] += red[256 + threadIdx.x + s];
    }
    __syncthreads();
  }
  if (threadIdx.x == 0) {
    float* part = (float*)(ws + WS_PARTIAL) + (size_t)k * 4096;
    part[blockIdx.x] = red[0];
    part[NBLOCKS + blockIdx.x] = red[256];
  }
}

// --- fold per-block partials, update sticky flag ---------------------------
__global__ void deq_reduce_kernel(unsigned char* __restrict__ ws, int k) {
  __shared__ float rd[512];
  const float* part = (const float*)(ws + WS_PARTIAL) + (size_t)k * 4096;
  const int t = threadIdx.x;
  float d2 = 0.0f, n2 = 0.0f;
  for (int i = t; i < NBLOCKS; i += 256) {
    d2 += part[i];
    n2 += part[NBLOCKS + i];
  }
  rd[t] = d2; rd[256 + t] = n2;
  __syncthreads();
  for (int s = 128; s > 0; s >>= 1) {
    if (t < s) { rd[t] += rd[t + s]; rd[256 + t] += rd[256 + t + s]; }
    __syncthreads();
  }
  if (t == 0) {
    int* flags = (int*)(ws + WS_FLAGS);
    const float res = sqrtf(rd[0]) / (sqrtf(rd[256]) + 1e-5f);
    flags[k] = flags[k - 1] | (res < EPS_RES ? 1 : 0);
  }
}

extern "C" void kernel_launch(void* const* d_in, const int* in_sizes, int n_in,
                              void* d_out, int out_size, void* d_ws, size_t ws_size,
                              hipStream_t stream) {
  const float* u  = (const float*)d_in[0];
  const float* Aw = (const float*)d_in[1];
  const float* Ab = (const float*)d_in[2];
  const float* Bw = (const float*)d_in[3];
  const float* Bb = (const float*)d_in[4];
  unsigned char* ws = (unsigned char*)d_ws;
  float* z = (float*)d_out;

  deq_prep_kernel<<<(DIM * DIM + 255) / 256, 256, 0, stream>>>(Aw, Ab, Bw, Bb, ws);
  deq_inject_kernel<<<NBLOCKS, 256, W_LDS_BYTES, stream>>>(u, ws, z);
  for (int k = 1; k < NITER; ++k) {
    deq_iter_kernel<<<NBLOCKS, 256, W_LDS_BYTES + RED_LDS_BYTES, stream>>>(ws, z, k);
    deq_reduce_kernel<<<1, 256, 0, stream>>>(ws, k);
  }
}
